// BitMoEFFN_16716012716238
// MI455X (gfx1250) — 8-bit matrix operand order repaired and checked by reading, compiled, not run on hardware
//
#include <hip/hip_runtime.h>
#include <cstdint>
#include <cstddef>

#define DEV __device__ __forceinline__

typedef __attribute__((ext_vector_type(8))) int v8i;

constexpr int kD    = 768;
constexpr int kFFN  = 3072;
constexpr int kBT   = 4096;
constexpr int kNE   = 8;
constexpr int kCap  = 640;
constexpr int kKeep = 1690;
constexpr int kBStride = 80;

DEV v8i wmma_iu8(v8i a, v8i b, v8i c) {
  return __builtin_amdgcn_wmma_i32_16x16x64_iu8(true, a, true, b, c, false, false);
}

DEV v8i v8i_zero() {
  v8i z;
#pragma unroll
  for (int i = 0; i < 8; ++i) z[i] = 0;
  return z;
}

DEV v8i load_a_i8(const signed char* __restrict__ A, int lda, int m0, int k0, int lane) {
  const int row  = m0 + (lane & 15);
  const int kofs = (lane & 16) ? 16 : 0;
  const int* p = reinterpret_cast<const int*>(A + (size_t)row * lda + k0 + kofs);
  v8i r;
  r[0] = p[0];  r[1] = p[1];
  r[2] = p[2];  r[3] = p[3];
  r[4] = p[8];  r[5] = p[9];
  r[6] = p[10]; r[7] = p[11];
  return r;
}

DEV v8i load_b_lds(const signed char* tile, int lane) {
  const int col  = lane & 15;
  const int kofs = (lane & 16) ? 16 : 0;
  const int* p = reinterpret_cast<const int*>(tile + col * kBStride + kofs);
  v8i r;
  r[0] = p[0]; r[1] = p[1]; r[2] = p[2];  r[3] = p[3];
  r[4] = p[8]; r[5] = p[9]; r[6] = p[10]; r[7] = p[11];
  return r;
}

DEV void prefetch_tile_async(const signed char* __restrict__ W, int ldw, int n0,
                             int k0, unsigned lds_base, int tid) {
  const int c   = tid >> 2;
  const int seg = tid & 3;
  const unsigned long long g =
      (unsigned long long)(uintptr_t)(W + (size_t)(n0 + c) * ldw + k0 + seg * 16);
  const unsigned l = lds_base + (unsigned)(c * kBStride + seg * 16);
  asm volatile("global_load_async_to_lds_b128 %0, %1, off"
               :: "v"(l), "v"(g) : "memory");
}

DEV void wait_async_all() {
  asm volatile("s_wait_asynccnt 0x0" ::: "memory");
}

__global__ void k_zero(float* __restrict__ p, long n) {
  long i = (long)blockIdx.x * blockDim.x + threadIdx.x;
  const long stride = (long)gridDim.x * blockDim.x;
  for (; i < n; i += stride) p[i] = 0.f;
}

__global__ __launch_bounds__(256)
void k_ternarize(const float* __restrict__ W, signed char* __restrict__ Wq,
                 float* __restrict__ scale, int nper) {
  __shared__ float red[256];
  const size_t base = (size_t)blockIdx.x * nper;
  float acc = 0.f;
  for (int i = threadIdx.x; i < nper; i += 256) acc += fabsf(W[base + i]);
  red[threadIdx.x] = acc;
  __syncthreads();
  for (int s = 128; s > 0; s >>= 1) {
    if (threadIdx.x < s) red[threadIdx.x] += red[threadIdx.x + s];
    __syncthreads();
  }
  const float sc = fmaxf(red[0] / (float)nper, 1e-8f);
  if (threadIdx.x == 0) scale[blockIdx.x] = sc;
  const float inv = 1.f / sc;
  for (int i = threadIdx.x; i < nper; i += 256) {
    const float q = fminf(fmaxf(rintf(W[base + i] * inv), -1.f), 1.f);
    Wq[base + i] = (signed char)(int)q;
  }
}

__global__ __launch_bounds__(256)
void k_quant_x(const float* __restrict__ X, signed char* __restrict__ Xq,
               float* __restrict__ sx) {
  __shared__ float red[256];
  const int row = blockIdx.x;
  const float* x = X + (size_t)row * kD;
  float m = 0.f;
  for (int i = threadIdx.x; i < kD; i += 256) m = fmaxf(m, fabsf(x[i]));
  red[threadIdx.x] = m;
  __syncthreads();
  for (int s = 128; s > 0; s >>= 1) {
    if (threadIdx.x < s) red[threadIdx.x] = fmaxf(red[threadIdx.x], red[threadIdx.x + s]);
    __syncthreads();
  }
  const float amax = fmaxf(red[0], 1e-8f);
  const float s = 7.f / amax;
  for (int i = threadIdx.x; i < kD; i += 256) {
    const float q = fminf(fmaxf(rintf(x[i] * s), -7.f), 7.f);
    Xq[(size_t)row * kD + i] = (signed char)(int)q;
  }
  if (threadIdx.x == 0) sx[row] = amax / 7.f;
}

__global__ __launch_bounds__(256)
void k_router_absmax(const float* __restrict__ W, float* __restrict__ s_out) {
  __shared__ float red[256];
  float m = 0.f;
  for (int i = threadIdx.x; i < kNE * kD; i += 256) m = fmaxf(m, fabsf(W[i]));
  red[threadIdx.x] = m;
  __syncthreads();
  for (int s = 128; s > 0; s >>= 1) {
    if (threadIdx.x < s) red[threadIdx.x] = fmaxf(red[threadIdx.x], red[threadIdx.x + s]);
    __syncthreads();
  }
  if (threadIdx.x == 0) s_out[0] = fmaxf(red[0], 1e-8f);
}

__global__ __launch_bounds__(256)
void k_router(const float* __restrict__ X, const float* __restrict__ RW,
              const float* __restrict__ s_amax, float* __restrict__ probs,
              int* __restrict__ idx2, float* __restrict__ p2,
              float* __restrict__ cw) {
  const int t = blockIdx.x * 256 + threadIdx.x;
  const float amax = s_amax[0];
  const float s = 127.f / amax;
  const float inv_s = amax / 127.f;
  const float* x = X + (size_t)t * kD;
  float logit[kNE];
#pragma unroll
  for (int e = 0; e < kNE; ++e) {
    const float* w = RW + e * kD;
    float acc = 0.f;
    for (int k = 0; k < kD; ++k) {
      const float wq = fminf(fmaxf(rintf(w[k] * s), -127.f), 127.f) * inv_s;
      acc += x[k] * wq;
    }
    logit[e] = acc;
  }
  float mx = logit[0];
#pragma unroll
  for (int e = 1; e < kNE; ++e) mx = fmaxf(mx, logit[e]);
  float p[kNE], den = 0.f;
#pragma unroll
  for (int e = 0; e < kNE; ++e) { p[e] = __expf(logit[e] - mx); den += p[e]; }
#pragma unroll
  for (int e = 0; e < kNE; ++e) {
    p[e] /= den;
    probs[(size_t)t * kNE + e] = p[e];
    cw[(size_t)t * kNE + e] = 0.f;
  }
  int i0 = 0;
#pragma unroll
  for (int e = 1; e < kNE; ++e) if (p[e] > p[i0]) i0 = e;
  int i1 = (i0 == 0) ? 1 : 0;
#pragma unroll
  for (int e = 0; e < kNE; ++e) if (e != i0 && p[e] > p[i1]) i1 = e;
  const float sum2 = p[i0] + p[i1];
  idx2[t * 2 + 0] = i0;  idx2[t * 2 + 1] = i1;
  p2[t * 2 + 0] = p[i0] / sum2;
  p2[t * 2 + 1] = p[i1] / sum2;
}

__global__ void k_capacity(const int* __restrict__ idx2, const float* __restrict__ p2,
                           float* __restrict__ cw, int* __restrict__ counts) {
  const int e = threadIdx.x;
  if (e >= kNE) return;
  int c = 0, tot = 0;
  for (int s = 0; s < kBT * 2; ++s) {
    if (idx2[s] == e) {
      ++tot; ++c;
      if (c <= kCap) cw[(size_t)(s >> 1) * kNE + e] = p2[s];
    }
  }
  counts[e] = tot;
}

__global__ __launch_bounds__(256)
void k_gateup(const signed char* __restrict__ Aq, const float* __restrict__ sA,
              const signed char* __restrict__ Wg, const signed char* __restrict__ Wu,
              const float* __restrict__ sWg, const float* __restrict__ sWu,
              int e, float* __restrict__ H) {
  __shared__ __align__(128) signed char sB[2][2][64 * kBStride];
  const int tid  = threadIdx.x;
  const int lane = tid & 31;
  const int wave = tid >> 5;
  const int m0 = blockIdx.y * 128 + wave * 16;
  const int n0 = blockIdx.x * 64;
  const float sg = sWg[e], su = sWu[e];

  unsigned lbase[2][2];
#pragma unroll
  for (int b = 0; b < 2; ++b)
#pragma unroll
    for (int mtx = 0; mtx < 2; ++mtx)
      lbase[b][mtx] = (unsigned)(uintptr_t)&sB[b][mtx][0];

  v8i cg[4], cu[4];
#pragma unroll
  for (int j = 0; j < 4; ++j) { cg[j] = v8i_zero(); cu[j] = v8i_zero(); }

  prefetch_tile_async(Wg, kD, n0, 0, lbase[0][0], tid);
  prefetch_tile_async(Wu, kD, n0, 0, lbase[0][1], tid);

  int buf = 0;
  for (int ks = 0; ks < kD / 64; ++ks) {
    wait_async_all();
    __syncthreads();
    const int k = ks * 64;
    if (ks + 1 < kD / 64) {
      prefetch_tile_async(Wg, kD, n0, k + 64, lbase[buf ^ 1][0], tid);
      prefetch_tile_async(Wu, kD, n0, k + 64, lbase[buf ^ 1][1], tid);
    }
    const v8i a = load_a_i8(Aq, kD, m0, k, lane);
#pragma unroll
    for (int j = 0; j < 4; ++j) {
      const v8i bg = load_b_lds(&sB[buf][0][16 * j * kBStride], lane);
      cg[j] = wmma_iu8(a, bg, cg[j]);
      const v8i bu = load_b_lds(&sB[buf][1][16 * j * kBStride], lane);
      cu[j] = wmma_iu8(a, bu, cu[j]);
    }
    buf ^= 1;
  }

  const int col0  = n0 + (lane & 15);
  const int rhalf = (lane & 16) ? 8 : 0;
#pragma unroll
  for (int v = 0; v < 8; ++v) {
    const int row = m0 + rhalf + v;
    const float srow = sA[row];
#pragma unroll
    for (int j = 0; j < 4; ++j) {
      const float g = (float)cg[j][v] * srow * sg;
      const float u = (float)cu[j][v] * srow * su;
      const float h = g / (1.f + __expf(-g)) * u;
      H[(size_t)row * kFFN + col0 + 16 * j] = h;
    }
  }
}

__global__ __launch_bounds__(256)
void k_topk_quant(const float* __restrict__ H, signed char* __restrict__ Hq,
                  float* __restrict__ sh) {
  __shared__ float ax[kFFN];
  __shared__ float red[256];
  __shared__ int cnt;
  const int row = blockIdx.x;
  const float* h = H + (size_t)row * kFFN;
  float m = 0.f;
  for (int i = threadIdx.x; i < kFFN; i += 256) {
    const float a = fabsf(h[i]);
    ax[i] = a;
    m = fmaxf(m, a);
  }
  red[threadIdx.x] = m;
  __syncthreads();
  for (int s = 128; s > 0; s >>= 1) {
    if (threadIdx.x < s) red[threadIdx.x] = fmaxf(red[threadIdx.x], red[threadIdx.x + s]);
    __syncthreads();
  }
  const float amax = fmaxf(red[0], 1e-8f);

  unsigned prefix = 0;
  for (int bit = 31; bit >= 0; --bit) {
    const unsigned cand = prefix | (1u << bit);
    if (threadIdx.x == 0) cnt = 0;
    __syncthreads();
    int c = 0;
    for (int i = threadIdx.x; i < kFFN; i += 256)
      c += (__float_as_uint(ax[i]) >= cand) ? 1 : 0;
    atomicAdd(&cnt, c);
    __syncthreads();
    if (cnt >= kKeep) prefix = cand;
    __syncthreads();
  }
  const float thresh = __uint_as_float(prefix);
  const float s = 127.f / amax;
  for (int i = threadIdx.x; i < kFFN; i += 256) {
    float q = 0.f;
    if (ax[i] >= thresh) q = fminf(fmaxf(rintf(h[i] * s), -127.f), 127.f);
    Hq[(size_t)row * kFFN + i] = (signed char)(int)q;
  }
  if (threadIdx.x == 0) sh[row] = amax / 127.f;
}

__global__ __launch_bounds__(256)
void k_down(const signed char* __restrict__ Hq, const float* __restrict__ sh,
            const signed char* __restrict__ Wd, const float* __restrict__ sWd,
            int e, const float* __restrict__ cw, float* __restrict__ out) {
  __shared__ __align__(128) signed char sB[2][64 * kBStride];
  const int tid  = threadIdx.x;
  const int lane = tid & 31;
  const int wave = tid >> 5;
  const int m0 = blockIdx.y * 128 + wave * 16;
  const int n0 = blockIdx.x * 64;
  const float sd = sWd[e];

  unsigned lbase[2];
  lbase[0] = (unsigned)(uintptr_t)&sB[0][0];
  lbase[1] = (unsigned)(uintptr_t)&sB[1][0];

  v8i c[4];
#pragma unroll
  for (int j = 0; j < 4; ++j) c[j] = v8i_zero();

  prefetch_tile_async(Wd, kFFN, n0, 0, lbase[0], tid);

  int buf = 0;
  for (int ks = 0; ks < kFFN / 64; ++ks) {
    wait_async_all();
    __syncthreads();
    const int k = ks * 64;
    if (ks + 1 < kFFN / 64)
      prefetch_tile_async(Wd, kFFN, n0, k + 64, lbase[buf ^ 1], tid);
    const v8i a = load_a_i8(Hq, kFFN, m0, k, lane);
#pragma unroll
    for (int j = 0; j < 4; ++j) {
      const v8i b = load_b_lds(&sB[buf][16 * j * kBStride], lane);
      c[j] = wmma_iu8(a, b, c[j]);
    }
    buf ^= 1;
  }

  const int col0  = n0 + (lane & 15);
  const int rhalf = (lane & 16) ? 8 : 0;
#pragma unroll
  for (int v = 0; v < 8; ++v) {
    const int row = m0 + rhalf + v;
    const float scale = sh[row] * sd;
    const float w = cw[(size_t)row * kNE + e];
#pragma unroll
    for (int j = 0; j < 4; ++j) {
      const float val = (float)c[j][v] * scale;
      out[(size_t)row * kD + col0 + 16 * j] += w * val;
    }
  }
}

__global__ __launch_bounds__(256)
void k_aux(const float* __restrict__ probs, const int* __restrict__ counts,
           float* __restrict__ out_aux) {
  __shared__ float red[kNE][256];
  float ps[kNE];
#pragma unroll
  for (int e = 0; e < kNE; ++e) ps[e] = 0.f;
  for (int t = threadIdx.x; t < kBT; t += 256)
#pragma unroll
    for (int e = 0; e < kNE; ++e) ps[e] += probs[(size_t)t * kNE + e];
#pragma unroll
  for (int e = 0; e < kNE; ++e) red[e][threadIdx.x] = ps[e];
  __syncthreads();
  for (int s = 128; s > 0; s >>= 1) {
    if (threadIdx.x < s)
#pragma unroll
      for (int e = 0; e < kNE; ++e) red[e][threadIdx.x] += red[e][threadIdx.x + s];
    __syncthreads();
  }
  if (threadIdx.x == 0) {
    float aux = 0.f;
#pragma unroll
    for (int e = 0; e < kNE; ++e) {
      const float frac  = (float)counts[e] / (float)(kBT * 2);
      const float pmean = red[e][0] / (float)kBT;
      aux += frac * pmean;
    }
    out_aux[0] = (float)kNE * aux;
  }
}

extern "C" void kernel_launch(void* const* d_in, const int* in_sizes, int n_in,
                              void* d_out, int out_size, void* d_ws, size_t ws_size,
                              hipStream_t stream) {
  (void)in_sizes; (void)n_in; (void)ws_size;
  const float* x  = (const float*)d_in[0];
  const float* gw = (const float*)d_in[1];
  const float* uw = (const float*)d_in[2];
  const float* dw = (const float*)d_in[3];
  const float* rw = (const float*)d_in[4];
  float* out = (float*)d_out;

  char* ws = (char*)d_ws;
  size_t off = 0;
  auto carve = [&](size_t bytes) -> char* {
    char* p = ws + off;
    off += (bytes + 255) & ~(size_t)255;
    return p;
  };
  const size_t wsz = (size_t)kFFN * kD;
  signed char* wq_g = (signed char*)carve(kNE * wsz);
  signed char* wq_u = (signed char*)carve(kNE * wsz);
  signed char* wq_d = (signed char*)carve(kNE * wsz);
  float* s_g = (float*)carve(kNE * 4);
  float* s_u = (float*)carve(kNE * 4);
  float* s_d = (float*)carve(kNE * 4);
  float* s_r = (float*)carve(4);
  signed char* xq = (signed char*)carve((size_t)kBT * kD);
  float* sx    = (float*)carve(kBT * 4);
  float* probs = (float*)carve((size_t)kBT * kNE * 4);
  float* cw    = (float*)carve((size_t)kBT * kNE * 4);
  int*   idx2  = (int*)carve((size_t)kBT * 2 * 4);
  float* p2    = (float*)carve((size_t)kBT * 2 * 4);
  int*   counts = (int*)carve(kNE * 4);
  float* H  = (float*)carve((size_t)kBT * kFFN * 4);
  signed char* Hq = (signed char*)carve((size_t)kBT * kFFN);
  float* sh = (float*)carve(kBT * 4);

  k_zero<<<1024, 256, 0, stream>>>(out, (long)out_size);
  k_ternarize<<<kNE, 256, 0, stream>>>(gw, wq_g, s_g, kFFN * kD);
  k_ternarize<<<kNE, 256, 0, stream>>>(uw, wq_u, s_u, kFFN * kD);
  k_ternarize<<<kNE, 256, 0, stream>>>(dw, wq_d, s_d, kD * kFFN);
  k_router_absmax<<<1, 256, 0, stream>>>(rw, s_r);
  k_quant_x<<<kBT, 256, 0, stream>>>(x, xq, sx);
  k_router<<<kBT / 256, 256, 0, stream>>>(x, rw, s_r, probs, idx2, p2, cw);
  k_capacity<<<1, 32, 0, stream>>>(idx2, p2, cw, counts);

  for (int e = 0; e < kNE; ++e) {
    k_gateup<<<dim3(kFFN / 64, kBT / 128), 256, 0, stream>>>(
        xq, sx, wq_g + e * wsz, wq_u + e * wsz, s_g, s_u, e, H);
    k_topk_quant<<<kBT, 256, 0, stream>>>(H, Hq, sh);
    k_down<<<dim3(kD / 64, kBT / 128), 256, 0, stream>>>(
        Hq, sh, wq_d + e * wsz, s_d, e, cw, out);
  }
  k_aux<<<1, 256, 0, stream>>>(probs, counts, out + (size_t)kBT * kD);
}
